// Transformers_14525579395529
// MI455X (gfx1250) — compile-verified, matrix operand order checked by reading, not run on hardware
//
#include <hip/hip_runtime.h>
#include <hip/hip_bf16.h>
#include <math.h>

// ---------------------------------------------------------------------------
// CDNA5 (gfx1250) implementation: 3-branch transformer + attention pooling.
// GEMMs: V_WMMA_F32_16X16X32_F16, 64x128 block tiles, float2 global staging,
// b128 LDS fragment loads, reg->LDS pipeline, global_prefetch of K+2 tile.
// MHA: flash-style streaming kernel (no score materialization).
// Custom column-softmax attention exploits symmetry of y@y^T.
// ---------------------------------------------------------------------------

typedef _Float16 half_t;
typedef __attribute__((ext_vector_type(16))) _Float16 v16h;
typedef __attribute__((ext_vector_type(8)))  _Float16 h8;
typedef __attribute__((ext_vector_type(2)))  _Float16 h2;
typedef __attribute__((ext_vector_type(8)))  float    v8f;

__device__ __forceinline__ v8f wmma_f16(v16h a, v16h b, v8f c) {
  // 8 args: (neg_a, A, neg_b, B, c_mod, C, reuse_a, reuse_b)
  return __builtin_amdgcn_wmma_f32_16x16x32_f16(false, a, false, b, (short)0, c,
                                                false, false);
}

// A-style 16x32 f16 fragment from LDS: rows = M (or N for the W^T operand),
// inner dim = K.  Per ISA 7.12.2 each lane's 16 halves are two contiguous
// 16-byte chunks: K in [8g,8g+8) and [16+8g,16+8g+8).  With a 16B-aligned
// row pitch this is exactly two ds_load_b128.
__device__ __forceinline__ v16h frag_row(const half_t* __restrict__ lds,
                                         int row0, int ld, int k0) {
  const int lane = threadIdx.x & 31;
  const int r = lane & 15;
  const int g = lane >> 4;
  const half_t* p = lds + (size_t)(row0 + r) * ld + k0 + 8 * g;
  h8 lo = *(const h8*)p;          // K = k0+8g   .. +7
  h8 hi = *(const h8*)(p + 16);   // K = k0+16+8g.. +7
  return __builtin_shufflevector(lo, hi, 0, 1, 2, 3, 4, 5, 6, 7, 8, 9, 10, 11,
                                 12, 13, 14, 15);
}

// Fragment from an LDS tile stored [K][N] (V tiles in attention): element
// (n0+r, k) read as lds[k*ld + n0 + r] (strided; scalar reads).
__device__ __forceinline__ v16h frag_col(const half_t* __restrict__ lds,
                                         int n0, int ld, int k0) {
  const int lane = threadIdx.x & 31;
  const int r = lane & 15;
  const int g = lane >> 4;
  v16h f;
#pragma unroll
  for (int v = 0; v < 8; ++v) {
    int k = ((v < 4) ? (2 * v) : (16 + 2 * (v - 4))) + 8 * g + k0;
    f[2 * v]     = lds[(size_t)k * ld + n0 + r];
    f[2 * v + 1] = lds[(size_t)(k + 1) * ld + n0 + r];
  }
  return f;
}

enum { GF_BIAS = 1, GF_RELU = 2, GF_ACCUM = 4, GF_TRANSB = 8 };

// ---------------------------------------------------------------------------
// Generic batched GEMM:  out = act(A[M,K] @ op(B) + bias) (+= if ACCUM)
//   default: Bsrc is W[N,K] row-major (computes A @ W^T)
//   TRANSB : Bsrc is B[K,N] row-major (computes A @ B)
// 64x128x32 block tiles, 8 waves; each wave computes 16x64 (4 WMMAs/K-step).
// Requires lda, ldb even (all call sites satisfy this; score buffer padded).
// ---------------------------------------------------------------------------
__global__ __launch_bounds__(256) void gemm_wmma_kernel(
    const float* __restrict__ A, const float* __restrict__ Bsrc,
    const float* __restrict__ bias, float* __restrict__ out,
    int M, int N, int K, int lda, int ldb, int ldo,
    long long strideA, long long strideB, long long strideO, int flags) {
  constexpr int BM = 64, BN = 128, BK = 32, LDT = 40;  // 16B-aligned row pitch
  __shared__ alignas(16) half_t sA[BM * LDT];
  __shared__ alignas(16) half_t sB[BN * LDT];
  A    += (size_t)blockIdx.z * strideA;
  Bsrc += (size_t)blockIdx.z * strideB;
  out  += (size_t)blockIdx.z * strideO;
  const int tid = threadIdx.x;
  const int wid = tid >> 5;
  const int m0 = blockIdx.y * BM;
  const int n0 = blockIdx.x * BN;
  const int wrow = (wid >> 1) * 16;   // 0,16,32,48
  const int wcol = (wid & 1) * 64;    // 0,64
  const bool transb = (flags & GF_TRANSB) != 0;
  const bool interior = (m0 + BM <= M) && (n0 + BN <= N);
  const int nk = (K + BK - 1) / BK;

  float2 ra[4];
  float2 rb[8];
  // Loads tile t into registers; returns true if rb[] holds the TRANSB
  // (n-pair, k) layout, false for the (row, col-pair) layout.
  auto load_regs = [&](int t) -> bool {
    const int k0 = t * BK;
    if (interior && (k0 + BK <= K)) {
#pragma unroll
      for (int i = 0; i < 4; ++i) {                 // A: 64x32 = 1024 pairs
        int idx = tid + i * 256;
        int r = idx >> 4, c2 = (idx & 15) * 2;
        ra[i] = *(const float2*)&A[(size_t)(m0 + r) * lda + k0 + c2];
      }
      if (transb) {
#pragma unroll
        for (int i = 0; i < 8; ++i) {               // pairs along contiguous N
          int idx = tid + i * 256;
          int n2 = (idx & 63) * 2, k = idx >> 6;
          rb[i] = *(const float2*)&Bsrc[(size_t)(k0 + k) * ldb + n0 + n2];
        }
        return true;
      }
#pragma unroll
      for (int i = 0; i < 8; ++i) {                 // B: 128x32 = 2048 pairs
        int idx = tid + i * 256;
        int r = idx >> 4, c2 = (idx & 15) * 2;
        rb[i] = *(const float2*)&Bsrc[(size_t)(n0 + r) * ldb + k0 + c2];
      }
      return false;
    }
    // guarded scalar path (boundary blocks / K tail)
#pragma unroll
    for (int i = 0; i < 4; ++i) {
      int idx = tid + i * 256;
      int r = idx >> 4, c2 = (idx & 15) * 2;
      int gm = m0 + r, gk = k0 + c2;
      ra[i].x = (gm < M && gk < K) ? A[(size_t)gm * lda + gk] : 0.f;
      ra[i].y = (gm < M && gk + 1 < K) ? A[(size_t)gm * lda + gk + 1] : 0.f;
    }
#pragma unroll
    for (int i = 0; i < 8; ++i) {
      int idx = tid + i * 256;
      int r = idx >> 4, c2 = (idx & 15) * 2;
      int gn = n0 + r, gk = k0 + c2;
      float vx = 0.f, vy = 0.f;
      if (gn < N && gk < K)
        vx = transb ? Bsrc[(size_t)gk * ldb + gn] : Bsrc[(size_t)gn * ldb + gk];
      if (gn < N && gk + 1 < K)
        vy = transb ? Bsrc[(size_t)(gk + 1) * ldb + gn]
                    : Bsrc[(size_t)gn * ldb + gk + 1];
      rb[i].x = vx; rb[i].y = vy;
    }
    return false;
  };

  v8f acc[4];
#pragma unroll
  for (int t = 0; t < 4; ++t)
#pragma unroll
    for (int v = 0; v < 8; ++v) acc[t][v] = 0.f;

  bool tileT = load_regs(0);
  for (int t = 0; t < nk; ++t) {
#pragma unroll
    for (int i = 0; i < 4; ++i) {           // A regs -> LDS (packed h2)
      int idx = tid + i * 256;
      int r = idx >> 4, c2 = (idx & 15) * 2;
      *(h2*)&sA[r * LDT + c2] = (h2){(half_t)ra[i].x, (half_t)ra[i].y};
    }
    if (tileT) {
#pragma unroll
      for (int i = 0; i < 8; ++i) {         // scatter transposed pairs
        int idx = tid + i * 256;
        int n2 = (idx & 63) * 2, k = idx >> 6;
        sB[n2 * LDT + k]       = (half_t)rb[i].x;
        sB[(n2 + 1) * LDT + k] = (half_t)rb[i].y;
      }
    } else {
#pragma unroll
      for (int i = 0; i < 8; ++i) {
        int idx = tid + i * 256;
        int r = idx >> 4, c2 = (idx & 15) * 2;
        *(h2*)&sB[r * LDT + c2] = (h2){(half_t)rb[i].x, (half_t)rb[i].y};
      }
    }
    __syncthreads();
    bool nextT = tileT;
    if (t + 1 < nk) nextT = load_regs(t + 1);  // overlap loads with WMMA
    if (interior && (t + 2) * BK + BK <= K) {
      // speculative prefetch of tile t+2 (global_prefetch_b8)
      __builtin_prefetch(
          &A[(size_t)(m0 + (tid >> 5)) * lda + (t + 2) * BK + (tid & 31)], 0, 3);
      __builtin_prefetch(
          transb ? &Bsrc[(size_t)((t + 2) * BK + (tid & 31)) * ldb + n0 + (tid >> 5)]
                 : &Bsrc[(size_t)(n0 + (tid >> 5)) * ldb + (t + 2) * BK + (tid & 31)],
          0, 3);
    }
    v16h af = frag_row(sA, wrow, LDT, 0);
#pragma unroll
    for (int tt = 0; tt < 4; ++tt) {
      v16h bf = frag_row(sB, wcol + tt * 16, LDT, 0);
      acc[tt] = wmma_f16(af, bf, acc[tt]);
    }
    __syncthreads();
    tileT = nextT;
  }
  // Epilogue. C layout (ISA 7.12.2): lane col = lane&15, rows 8*(lane>>4)+v.
  const int lane = tid & 31;
  const int col = lane & 15;
  const int rbase = (lane >> 4) * 8;
#pragma unroll
  for (int tt = 0; tt < 4; ++tt) {
    int gn = n0 + wcol + tt * 16 + col;
    if (gn >= N) continue;
    float bv = (flags & GF_BIAS) ? bias[gn] : 0.f;
#pragma unroll
    for (int v = 0; v < 8; ++v) {
      int gm = m0 + wrow + rbase + v;
      if (gm >= M) continue;
      float r = acc[tt][v] + bv;
      if (flags & GF_RELU) r = fmaxf(r, 0.f);
      size_t o = (size_t)gm * ldo + gn;
      out[o] = (flags & GF_ACCUM) ? (out[o] + r) : r;
    }
  }
}

// ---------------------------------------------------------------------------
// Flash-style MHA: one wave per 16-query tile, streams 32-key tiles with
// online softmax; S and P@V products via WMMA.  dh in {64,76,96} (mult of 4).
// qkv layout: [B, S, 3D]; q at col h*dh, k at D+h*dh, v at 2D+h*dh.
// out: [B, S, D] (heads re-interleaved).
// ---------------------------------------------------------------------------
#define DHMAX 96
#define KLD   104   // 16B-aligned row pitch (frag_row does ds_load_b128)
#define PLD   40

__global__ __launch_bounds__(64) void mha_flash_kernel(
    const float* __restrict__ qkv, float* __restrict__ out,
    int B, int S, int D, int H, int dh, float scale) {
  __shared__ alignas(16) half_t sQ[2][16 * KLD];
  __shared__ alignas(16) half_t sK[2][32 * KLD];
  __shared__ alignas(16) half_t sV[2][32 * KLD];
  __shared__ alignas(16) half_t sP[2][16 * PLD];
  const int wv = threadIdx.x >> 5;
  const int lane = threadIdx.x & 31;
  const int bh = blockIdx.y;
  const int b = bh / H, h = bh % H;
  const int qt = blockIdx.x * 2 + wv;
  if (qt * 16 >= S) return;                 // uniform per wave
  const int mq = qt * 16;
  half_t* Q  = sQ[wv];
  half_t* Kt = sK[wv];
  half_t* Vt = sV[wv];
  half_t* P  = sP[wv];
  const size_t rs3d = (size_t)3 * D;
  const float* base = qkv + (size_t)b * S * rs3d + h * dh;

  // stage Q (pre-scaled): lane -> (row, column-half); float4 loads, h2 stores
  {
    int r = lane >> 1;
    int cs = (lane & 1) * 48;
    bool rowok = (mq + r) < S;
    int lim = rowok ? ((dh < cs + 48) ? dh : (cs + 48)) : cs;  // mult of 4
    const float* rp = base + (size_t)(mq + r) * rs3d;
    int c = cs;
    for (; c + 3 < lim; c += 4) {
      float4 q4 = *(const float4*)(rp + c);
      *(h2*)&Q[r * KLD + c]     = (h2){(half_t)(q4.x * scale), (half_t)(q4.y * scale)};
      *(h2*)&Q[r * KLD + c + 2] = (h2){(half_t)(q4.z * scale), (half_t)(q4.w * scale)};
    }
    for (; c < cs + 48; c += 2)
      *(h2*)&Q[r * KLD + c] = (h2){(half_t)0.f, (half_t)0.f};
  }

  const int nchunk = (dh + 15) >> 4;        // O chunks of 16 cols
  const int nkk = (dh + 31) >> 5;           // dh chunks of 32 for scores
  float mrow[8], lrow[8], corr[8];
  v8f accO[6];
#pragma unroll
  for (int v = 0; v < 8; ++v) { mrow[v] = -3.0e38f; lrow[v] = 0.f; }
#pragma unroll
  for (int c = 0; c < 6; ++c)
#pragma unroll
    for (int v = 0; v < 8; ++v) accO[c][v] = 0.f;

  const int col = lane & 15;
  const int rbase = (lane >> 4) * 8;

  for (int kt0 = 0; kt0 < S; kt0 += 32) {
    // stage K and V: one lane per key row; float4 loads, h2 stores
    {
      int j = kt0 + lane;
      int lim = (j < S) ? dh : 0;           // dh mult of 4
      const float* rp = base + (size_t)j * rs3d;
      int c = 0;
      for (; c + 3 < lim; c += 4) {
        float4 kv = *(const float4*)(rp + D + c);
        float4 vv = *(const float4*)(rp + 2 * D + c);
        *(h2*)&Kt[lane * KLD + c]     = (h2){(half_t)kv.x, (half_t)kv.y};
        *(h2*)&Kt[lane * KLD + c + 2] = (h2){(half_t)kv.z, (half_t)kv.w};
        *(h2*)&Vt[lane * KLD + c]     = (h2){(half_t)vv.x, (half_t)vv.y};
        *(h2*)&Vt[lane * KLD + c + 2] = (h2){(half_t)vv.z, (half_t)vv.w};
      }
      for (; c < DHMAX; c += 2) {
        *(h2*)&Kt[lane * KLD + c] = (h2){(half_t)0.f, (half_t)0.f};
        *(h2*)&Vt[lane * KLD + c] = (h2){(half_t)0.f, (half_t)0.f};
      }
    }
    v8f s0 = {};
    v8f s1 = {};
    for (int kk = 0; kk < nkk; ++kk) {
      v16h a  = frag_row(Q, 0, KLD, kk * 32);
      v16h b0 = frag_row(Kt, 0, KLD, kk * 32);
      v16h b1 = frag_row(Kt, 16, KLD, kk * 32);
      s0 = wmma_f16(a, b0, s0);
      s1 = wmma_f16(a, b1, s1);
    }
    const int j0 = kt0 + col, j1 = kt0 + 16 + col;
#pragma unroll
    for (int v = 0; v < 8; ++v) {           // online softmax per query row
      float v0 = (j0 < S) ? s0[v] : -3.0e38f;
      float v1 = (j1 < S) ? s1[v] : -3.0e38f;
      float mx = fmaxf(v0, v1);
#pragma unroll
      for (int off = 1; off < 16; off <<= 1) mx = fmaxf(mx, __shfl_xor(mx, off, 32));
      float mnew = fmaxf(mrow[v], mx);
      float p0 = __expf(v0 - mnew);
      float p1 = __expf(v1 - mnew);
      float rsum = p0 + p1;
#pragma unroll
      for (int off = 1; off < 16; off <<= 1) rsum += __shfl_xor(rsum, off, 32);
      corr[v] = __expf(mrow[v] - mnew);
      lrow[v] = lrow[v] * corr[v] + rsum;
      mrow[v] = mnew;
      P[(rbase + v) * PLD + col]      = (half_t)p0;   // P tile 16x32, A-major
      P[(rbase + v) * PLD + 16 + col] = (half_t)p1;
    }
#pragma unroll
    for (int c = 0; c < 6; ++c)
#pragma unroll
      for (int v = 0; v < 8; ++v) accO[c][v] *= corr[v];
    v16h ap = frag_row(P, 0, PLD, 0);
    for (int c = 0; c < nchunk; ++c) {
      v16h bv = frag_col(Vt, c * 16, KLD, 0);          // V stored [k][d]
      accO[c] = wmma_f16(ap, bv, accO[c]);
    }
  }
  float* obase = out + (size_t)b * S * D + h * dh;
  for (int c = 0; c < nchunk; ++c) {
    int d = c * 16 + col;
    if (d >= dh) continue;
#pragma unroll
    for (int v = 0; v < 8; ++v) {
      int q = mq + rbase + v;
      if (q < S) obase[(size_t)q * D + d] = accO[c][v] / lrow[v];
    }
  }
}

// ---------------------------------------------------------------------------
// Elementwise / reduction helpers (fp32)
// ---------------------------------------------------------------------------
__global__ void add_pos_kernel(const float* __restrict__ x,
                               const float* __restrict__ pos,
                               float* __restrict__ out, long long SD,
                               long long total) {
  long long i = (long long)blockIdx.x * 256 + threadIdx.x;
  if (i < total) out[i] = x[i] + pos[i % SD];
}

__global__ __launch_bounds__(256) void layernorm_kernel(
    const float* __restrict__ in, const float* __restrict__ g,
    const float* __restrict__ bta, float* __restrict__ out, int D) {
  const size_t row = blockIdx.x;
  const float* x = in + row * D;
  float s = 0.f, s2 = 0.f;
  for (int i = threadIdx.x; i < D; i += 256) { float v = x[i]; s += v; s2 += v * v; }
  __shared__ float r1[256], r2[256];
  r1[threadIdx.x] = s; r2[threadIdx.x] = s2;
  __syncthreads();
  for (int st = 128; st > 0; st >>= 1) {
    if (threadIdx.x < st) { r1[threadIdx.x] += r1[threadIdx.x + st];
                            r2[threadIdx.x] += r2[threadIdx.x + st]; }
    __syncthreads();
  }
  float mu = r1[0] / D;
  float var = r2[0] / D - mu * mu;          // biased var, matches jnp.var
  float inv = rsqrtf(var + 1e-5f);
  for (int i = threadIdx.x; i < D; i += 256)
    out[row * D + i] = (x[i] - mu) * inv * g[i] + bta[i];
}

// Row max / sum-exp of s[b,i,:] (row pitch Sp) -- by symmetry of y@y^T these
// are the column stats needed for softmax(axis=1).
__global__ __launch_bounds__(256) void rowstats_kernel(
    const float* __restrict__ s, float* __restrict__ m, float* __restrict__ c,
    int S, int Sp) {
  const int row = blockIdx.x;               // b*S + i
  const int b = row / S;
  const float* x = s + (size_t)b * S * Sp + (size_t)(row - b * S) * Sp;
  __shared__ float red[256];
  float mx = -3.0e38f;
  for (int j = threadIdx.x; j < S; j += 256) mx = fmaxf(mx, x[j]);
  red[threadIdx.x] = mx;
  __syncthreads();
  for (int st = 128; st > 0; st >>= 1) {
    if (threadIdx.x < st) red[threadIdx.x] = fmaxf(red[threadIdx.x], red[threadIdx.x + st]);
    __syncthreads();
  }
  mx = red[0];
  __syncthreads();
  float sum = 0.f;
  for (int j = threadIdx.x; j < S; j += 256) sum += __expf(x[j] - mx);
  red[threadIdx.x] = sum;
  __syncthreads();
  for (int st = 128; st > 0; st >>= 1) {
    if (threadIdx.x < st) red[threadIdx.x] += red[threadIdx.x + st];
    __syncthreads();
  }
  if (threadIdx.x == 0) { m[row] = mx; c[row] = red[0]; }
}

// In-place column-normalize: s[b,i,j] = exp(s - m[b,j]) / c[b,j] (pitch Sp)
__global__ void colnorm_kernel(float* __restrict__ s, const float* __restrict__ m,
                               const float* __restrict__ c, int S, int Sp,
                               long long total) {
  long long i = (long long)blockIdx.x * 256 + threadIdx.x;
  if (i >= total) return;
  long long per = (long long)S * Sp;
  int b = (int)(i / per);
  int j = (int)(i % Sp);
  if (j >= S) return;                       // pad column, never read
  s[i] = __expf(s[i] - m[b * S + j]) / c[b * S + j];
}

__global__ void maxpool_kernel(const float* __restrict__ y,
                               float* __restrict__ pooled, int S, int D) {
  int b = blockIdx.y;
  int d = blockIdx.x * 256 + threadIdx.x;
  if (d >= D) return;
  const float* p = y + (size_t)b * S * D + d;
  float mx = -3.0e38f;
  for (int i = 0; i < S; ++i) mx = fmaxf(mx, p[(size_t)i * D]);
  pooled[(size_t)b * D + d] = mx;
}

__global__ void fc13_kernel(const float* __restrict__ in,
                            const float* __restrict__ w,
                            const float* __restrict__ bvec,
                            float* __restrict__ out) {
  int t = blockIdx.x * 64 + threadIdx.x;
  if (t >= 64 * 3) return;
  int b = t / 3, o = t % 3;
  float s = bvec[o];
  for (int k = 0; k < 64; ++k) s += in[b * 64 + k] * w[o * 64 + k];
  out[t] = s;
}

// ---------------------------------------------------------------------------
// Host-side orchestration
// ---------------------------------------------------------------------------
static inline void gemm(const float* A, const float* Bsrc, const float* bias,
                        float* out, int M, int N, int K, int lda, int ldb,
                        int ldo, long long sA, long long sB, long long sO,
                        int batch, int flags, hipStream_t st) {
  dim3 g((N + 127) / 128, (M + 63) / 64, batch);
  gemm_wmma_kernel<<<g, 256, 0, st>>>(A, Bsrc, bias, out, M, N, K, lda, ldb,
                                      ldo, sA, sB, sO, flags);
}

struct Ws {
  float *bx, *bh, *bq, *bt, *bm, *bc, *pooled, *acc, *fc8o;
};

static void run_branch(const float* xin, const float* const* P, const float* fw,
                       const float* fb, int B, int S, int D, int H, int dff,
                       bool first, const Ws& w, hipStream_t st) {
  const float *pos = P[0], *w_in = P[1], *b_in = P[2], *w_out = P[3],
              *b_out = P[4], *ln1g = P[5], *ln1b = P[6], *ln2g = P[7],
              *ln2b = P[8], *w1 = P[9], *b1 = P[10], *w2 = P[11], *b2 = P[12],
              *lnfg = P[13], *lnfb = P[14];
  const long long BS = (long long)B * S;
  const long long BSD = BS * D;
  const int Sp = (S + 1) & ~1;              // even row pitch for score buffer
  // x = input + pos
  add_pos_kernel<<<(int)((BSD + 255) / 256), 256, 0, st>>>(xin, pos, w.bx,
                                                           (long long)S * D, BSD);
  // h = LN1(x)
  layernorm_kernel<<<(int)BS, 256, 0, st>>>(w.bx, ln1g, ln1b, w.bh, D);
  // qkv = h @ w_in^T + b_in
  gemm(w.bh, w_in, b_in, w.bq, (int)BS, 3 * D, D, D, D, 3 * D, 0, 0, 0, 1,
       GF_BIAS, st);
  // attn -> bh (flash)
  {
    int dh = D / H;
    int qtiles = (S + 15) / 16;
    dim3 g((qtiles + 1) / 2, B * H);
    mha_flash_kernel<<<g, 64, 0, st>>>(w.bq, w.bh, B, S, D, H, dh,
                                       1.f / sqrtf((float)dh));
  }
  // x += attn @ w_out^T + b_out
  gemm(w.bh, w_out, b_out, w.bx, (int)BS, D, D, D, D, D, 0, 0, 0, 1,
       GF_BIAS | GF_ACCUM, st);
  // h = LN2(x)
  layernorm_kernel<<<(int)BS, 256, 0, st>>>(w.bx, ln2g, ln2b, w.bh, D);
  // t = relu(h @ w1^T + b1)
  gemm(w.bh, w1, b1, w.bt, (int)BS, dff, D, D, D, dff, 0, 0, 0, 1,
       GF_BIAS | GF_RELU, st);
  // x += t @ w2^T + b2
  gemm(w.bt, w2, b2, w.bx, (int)BS, D, dff, dff, dff, D, 0, 0, 0, 1,
       GF_BIAS | GF_ACCUM, st);
  // y = LNf(x)
  layernorm_kernel<<<(int)BS, 256, 0, st>>>(w.bx, lnfg, lnfb, w.bh, D);
  // s = y @ y^T (batched, symmetric), row pitch Sp
  gemm(w.bh, w.bh, nullptr, w.bq, S, S, D, D, D, Sp, (long long)S * D,
       (long long)S * D, (long long)S * Sp, B, 0, st);
  // column softmax via row stats (symmetry)
  rowstats_kernel<<<(int)BS, 256, 0, st>>>(w.bq, w.bm, w.bc, S, Sp);
  {
    long long tot = (long long)B * S * Sp;
    colnorm_kernel<<<(int)((tot + 255) / 256), 256, 0, st>>>(w.bq, w.bm, w.bc,
                                                             S, Sp, tot);
  }
  // y2 = P @ y (batched, B operand non-transposed)
  gemm(w.bq, w.bh, nullptr, w.bt, S, D, S, Sp, D, D, (long long)S * Sp,
       (long long)S * D, (long long)S * D, B, GF_TRANSB, st);
  // pooled = max over sequence
  maxpool_kernel<<<dim3((D + 255) / 256, B), 256, 0, st>>>(w.bt, w.pooled, S, D);
  // acc (+)= pooled @ fw^T + fb
  gemm(w.pooled, fw, fb, w.acc, B, 256, D, D, D, 256, 0, 0, 0, 1,
       GF_BIAS | (first ? 0 : GF_ACCUM), st);
}

extern "C" void kernel_launch(void* const* d_in, const int* in_sizes, int n_in,
                              void* d_out, int out_size, void* d_ws,
                              size_t ws_size, hipStream_t stream) {
  (void)in_sizes; (void)n_in; (void)out_size; (void)ws_size;
  auto F = [&](int i) { return (const float*)d_in[i]; };
  const float* x1 = F(0);
  const float* x2 = F(1);
  const float* x3 = F(2);
  const float* T1[15]; const float* T2[15]; const float* T3[15];
  for (int i = 0; i < 15; ++i) { T1[i] = F(3 + i); T2[i] = F(18 + i); T3[i] = F(33 + i); }

  // Workspace carve (floats). Max across branches (branch 2 dominates):
  float* Wf = (float*)d_ws;
  size_t off = 0;
  auto carve = [&](size_t n) { float* p = Wf + off; off += n; return p; };
  const size_t MAXBSD = 64ull * 499 * 512;    // residual/ln/ffn buffers
  const size_t MAXBIG = 64ull * 499 * 1536;   // qkv / padded score buffer
  Ws w;
  w.bx = carve(MAXBSD);
  w.bh = carve(MAXBSD);
  w.bq = carve(MAXBIG);
  w.bt = carve(MAXBSD);
  w.bm = carve(64ull * 499);
  w.bc = carve(64ull * 499);
  w.pooled = carve(64ull * 768);
  w.acc = carve(64ull * 256);
  w.fc8o = carve(64ull * 64);

  run_branch(x1, T1, F(48), F(49), 64, 99, 380, 5, 190, true, w, stream);
  run_branch(x2, T2, F(50), F(51), 64, 499, 512, 8, 256, false, w, stream);
  run_branch(x3, T3, F(52), F(53), 64, 69, 768, 8, 384, false, w, stream);

  // head: relu(acc @ fc8^T + b8) then fc13
  gemm(w.acc, F(54), F(55), w.fc8o, 64, 64, 256, 256, 256, 64, 0, 0, 0, 1,
       GF_BIAS | GF_RELU, stream);
  fc13_kernel<<<3, 64, 0, stream>>>(w.fc8o, F(56), F(57), (float*)d_out);
}
